// Feature_RNNWithCells_27943057228462
// MI455X (gfx1250) — matrix operand order repaired and checked by reading, not compiled, not run on hardware
//
#include <hip/hip_runtime.h>
#include <cstdint>

// ---------------------------------------------------------------------------
// Problem constants (match the reference)
// ---------------------------------------------------------------------------
constexpr int VDIM  = 4096;   // hidden / vocab size
constexpr int DFEAT = 64;     // feature dim (GEMM K)
constexpr int NSTEP = 512;    // B*S sequential steps
constexpr int SLEN  = 256;    // S (outputs kept for last S steps)
constexpr int NB    = 256;    // persistent-kernel blocks
constexpr int BS    = 256;    // threads per block (8 wave32)

typedef _Float16 f16;
typedef __attribute__((ext_vector_type(16))) _Float16 v16h;
typedef __attribute__((ext_vector_type(8)))  float    v8f;

// ---------------------------------------------------------------------------
// Helpers
// ---------------------------------------------------------------------------
__device__ __forceinline__ unsigned short f2bf(float f) {
  unsigned u = __float_as_uint(f);
  unsigned r = (u + 0x7fffu + ((u >> 16) & 1u)) >> 16;   // round-to-nearest-even
  return (unsigned short)r;
}
__device__ __forceinline__ float bflo(unsigned u) { return __uint_as_float(u << 16); }
__device__ __forceinline__ float bfhi(unsigned u) { return __uint_as_float(u & 0xffff0000u); }

// CDNA5 async copy: 16B per lane, global -> LDS, tracked by ASYNCcnt.
__device__ __forceinline__ void async_vec_load(float* vecLds, const float* src, int tid) {
#pragma unroll
  for (int i = 0; i < 4; ++i) {
    int idx = (tid + i * BS) * 4;                 // 4 floats = 16B granule
    unsigned ldsOff = (unsigned)(unsigned long long)
        (__attribute__((address_space(3))) float*)(vecLds + idx);
    unsigned long long gaddr = (unsigned long long)(const void*)(src + idx);
    asm volatile("global_load_async_to_lds_b128 %0, %1, off"
                 :: "v"(ldsOff), "v"(gaddr) : "memory");
  }
}
__device__ __forceinline__ void wait_async() {
  asm volatile("s_wait_asynccnt 0" ::: "memory");
}

// Device-scope grid barrier (all NB blocks co-resident).
__device__ __forceinline__ void grid_sync(unsigned* cnt, unsigned* gen) {
  __threadfence();
  __syncthreads();
  if (threadIdx.x == 0) {
    unsigned g = __hip_atomic_load(gen, __ATOMIC_RELAXED, __HIP_MEMORY_SCOPE_AGENT);
    unsigned prev = __hip_atomic_fetch_add(cnt, 1u, __ATOMIC_ACQ_REL, __HIP_MEMORY_SCOPE_AGENT);
    if (prev == (unsigned)(NB - 1)) {
      __hip_atomic_store(cnt, 0u, __ATOMIC_RELAXED, __HIP_MEMORY_SCOPE_AGENT);
      __hip_atomic_fetch_add(gen, 1u, __ATOMIC_RELEASE, __HIP_MEMORY_SCOPE_AGENT);
    } else {
      while (__hip_atomic_load(gen, __ATOMIC_ACQUIRE, __HIP_MEMORY_SCOPE_AGENT) == g)
        __builtin_amdgcn_s_sleep(2);
    }
  }
  __syncthreads();
}

// One bf16 GEMV row: 32 lanes x 16 iters x 8 bf16 (16B) coalesced; fp32 acc.
__device__ __forceinline__ float gemv_row(const unsigned short* __restrict__ wrow,
                                          const float* __restrict__ hs, int lane) {
  float acc = 0.f;
  const uint4* w4 = (const uint4*)wrow;
#pragma unroll 4
  for (int it = 0; it < 16; ++it) {
    int k0 = it * 256 + lane * 8;
    uint4  d  = w4[it * 32 + lane];
    float4 hA = *(const float4*)(hs + k0);
    float4 hB = *(const float4*)(hs + k0 + 4);
    acc += bflo(d.x) * hA.x + bfhi(d.x) * hA.y + bflo(d.y) * hA.z + bfhi(d.y) * hA.w;
    acc += bflo(d.z) * hB.x + bfhi(d.z) * hB.y + bflo(d.w) * hB.z + bfhi(d.w) * hB.w;
  }
#pragma unroll
  for (int off = 16; off; off >>= 1) acc += __shfl_xor(acc, off, 32);
  return acc;
}

// ---------------------------------------------------------------------------
// Kernel 1: zero the scan state + barrier counters
// ---------------------------------------------------------------------------
__global__ void __launch_bounds__(BS)
init_state(float* h, unsigned* sync, float* blkmax, float* blksum) {
  int i = blockIdx.x * BS + threadIdx.x;
  if (i < VDIM) h[i] = 0.f;
  if (i < 2)    sync[i] = 0u;
  if (i < NB) { blkmax[i] = 0.f; blksum[i] = 0.f; }
}

// ---------------------------------------------------------------------------
// Kernel 2: fp32 -> bf16 weight conversion (halves bytes; 96MB fits L2)
// ---------------------------------------------------------------------------
__global__ void __launch_bounds__(BS)
f32_to_bf16(const float* __restrict__ src, unsigned short* __restrict__ dst, int n4) {
  int i = blockIdx.x * BS + threadIdx.x;
  if (i >= n4) return;
  float4 f = ((const float4*)src)[i];
  uint2 r;
  r.x = (unsigned)f2bf(f.x) | ((unsigned)f2bf(f.y) << 16);
  r.y = (unsigned)f2bf(f.z) | ((unsigned)f2bf(f.w) << 16);
  ((uint2*)dst)[i] = r;
}

// ---------------------------------------------------------------------------
// Kernel 3: x_proj[i,v] = features[ids[i],:] . W_ih[v,:] + b_ih[v]
// 512x4096x64 GEMM via V_WMMA_F32_16X16X32_F16 (one wave per 16x16 tile).
// A layout per ISA 16-bit A-matrix table; B layout lane=column, K ascending.
// ---------------------------------------------------------------------------
__global__ void __launch_bounds__(BS)
xproj_gemm(const int* __restrict__ ids, const float* __restrict__ feats,
           const float* __restrict__ W_ih, const float* __restrict__ b_ih,
           float* __restrict__ xproj) {
  int wave  = (blockIdx.x * BS + threadIdx.x) >> 5;   // 0..8191
  int lane  = threadIdx.x & 31;
  int tileM = wave >> 8;                               // 0..31  (512/16)
  int tileN = wave & 255;                              // 0..255 (4096/16)
  int m     = lane & 15;
  int khalf = lane >> 4;                               // 0 or 1

  int rowA = tileM * 16 + m;
  int id   = ids[rowA];
  int ncol = tileN * 16 + m;                           // B column for this lane

  v8f c = {};
#pragma unroll
  for (int kb = 0; kb < DFEAT; kb += 32) {
    v16h a, b;
    // A (16x32 f16): lane<16 holds K {kb..kb+7, kb+16..kb+23}; lane>=16 the other halves
    const float* fr = feats + id * DFEAT + kb + khalf * 8;
    // B (32x16 f16): lane holds column ncol, K {kb+khalf*8..+7, kb+16+khalf*8..+7} (same as A)
    const float* wr = W_ih + (size_t)ncol * DFEAT + kb + khalf * 8;
#pragma unroll
    for (int j = 0; j < 8; ++j) {
      a[j]     = (f16)fr[j];
      a[j + 8] = (f16)fr[16 + j];
    }
#pragma unroll
    for (int j = 0; j < 8; ++j) { b[j] = (f16)wr[j]; b[j + 8] = (f16)wr[16 + j]; }
    c = __builtin_amdgcn_wmma_f32_16x16x32_f16(false, a, false, b,
                                               (short)0, c, false, false);
  }
  // C/D layout: VGPR r -> lane<16: (M=r, N=lane) ; lane>=16: (M=8+r, N=lane-16)
  int   n0    = tileN * 16 + (lane & 15);
  int   mBase = tileM * 16 + khalf * 8;
  float bias  = b_ih[n0];
#pragma unroll
  for (int r = 0; r < 8; ++r)
    xproj[(size_t)(mBase + r) * VDIM + n0] = c[r] + bias;
}

// ---------------------------------------------------------------------------
// Kernel 4: persistent 512-step scan. 256 blocks; block b owns rows
// [16b,16b+16); wave w owns rows 16b+2w, 16b+2w+1. 4 grid barriers / step.
// ---------------------------------------------------------------------------
__global__ void __launch_bounds__(BS)
rnn_scan(const unsigned short* __restrict__ Whh,
         const unsigned short* __restrict__ Wo,
         const unsigned short* __restrict__ Wh,
         const float* __restrict__ bhh, const float* __restrict__ bo,
         const float* __restrict__ bh,
         const float* __restrict__ xproj, const float* __restrict__ rmult,
         float* h, float* st, float* o, float* hid,
         float* blkmax, float* blksum, unsigned* sync,
         float* out_seq, float* out_hid) {
  __shared__ float vec[VDIM];   // 16KB: current shared vector (h / st / o)
  __shared__ float red[16];     // per-row partials for block reductions
  __shared__ float sMS[2];      // softmax global max / sum

  const int tid  = threadIdx.x;
  const int lane = tid & 31;
  const int wv   = tid >> 5;
  const int blk  = blockIdx.x;
  unsigned* cnt = sync;
  unsigned* gen = sync + 1;
  const int row0 = blk * 16 + wv * 2;

  for (int t = 0; t < NSTEP; ++t) {
    // ---- phase 1: st = tanh(xp[t] + Whh.h + bhh) -------------------------
    async_vec_load(vec, h, tid);
    wait_async();
    __syncthreads();
#pragma unroll
    for (int r = 0; r < 2; ++r) {
      int v = row0 + r;
      float acc = gemv_row(Whh + (size_t)v * VDIM, vec, lane);
      if (lane == 0)
        st[v] = tanhf(xproj[(size_t)t * VDIM + v] + acc + bhh[v]);
    }
    grid_sync(cnt, gen);

    // ---- phase 2: o = Wo.st + bo ; per-block max(o) ----------------------
    async_vec_load(vec, st, tid);
    wait_async();
    __syncthreads();
#pragma unroll
    for (int r = 0; r < 2; ++r) {
      int v = row0 + r;
      float acc = gemv_row(Wo + (size_t)v * VDIM, vec, lane);
      if (lane == 0) {
        float val = acc + bo[v];
        o[v] = val;
        red[wv * 2 + r] = val;
      }
    }
    __syncthreads();
    if (tid == 0) {
      float m = red[0];
#pragma unroll
      for (int i = 1; i < 16; ++i) m = fmaxf(m, red[i]);
      blkmax[blk] = m;
    }
    grid_sync(cnt, gen);

    // ---- phase 3: hid = Wh.o + bh ; global max + partial exp-sums --------
    async_vec_load(vec, o, tid);
    if (wv == 0) {                         // global max while async copy flies
      float m = -3.4e38f;
      for (int i = lane; i < NB; i += 32) m = fmaxf(m, blkmax[i]);
#pragma unroll
      for (int off = 16; off; off >>= 1) m = fmaxf(m, __shfl_xor(m, off, 32));
      if (lane == 0) sMS[0] = m;
    }
    wait_async();
    __syncthreads();
    float M = sMS[0];
#pragma unroll
    for (int r = 0; r < 2; ++r) {
      int v = row0 + r;
      float acc = gemv_row(Wh + (size_t)v * VDIM, vec, lane);
      if (lane == 0) {
        hid[v] = acc + bh[v];
        red[wv * 2 + r] = expf(vec[v] - M);   // vec holds o
      }
    }
    __syncthreads();
    if (tid == 0) {
      float s = 0.f;
#pragma unroll
      for (int i = 0; i < 16; ++i) s += red[i];
      blksum[blk] = s;
    }
    grid_sync(cnt, gen);

    // ---- phase 4: softmax*rand, h update, outputs ------------------------
    if (wv == 0) {
      float s = 0.f;
      for (int i = lane; i < NB; i += 32) s += blksum[i];
#pragma unroll
      for (int off = 16; off; off >>= 1) s += __shfl_xor(s, off, 32);
      if (lane == 0) sMS[1] = s;
    }
    __syncthreads();
    if (tid < 16) {
      int v = blk * 16 + tid;
      float outv = expf(o[v] - sMS[0]) / sMS[1] * rmult[(size_t)t * VDIM + v];
      float hv   = hid[v];
      h[v] = outv + hv;
      if (t >= NSTEP - SLEN) out_seq[(size_t)(t - (NSTEP - SLEN)) * VDIM + v] = outv;
      if (t == NSTEP - 1)    out_hid[v] = hv;
    }
    grid_sync(cnt, gen);
  }
}

// ---------------------------------------------------------------------------
// Launch
// ---------------------------------------------------------------------------
extern "C" void kernel_launch(void* const* d_in, const int* in_sizes, int n_in,
                              void* d_out, int out_size, void* d_ws, size_t ws_size,
                              hipStream_t stream) {
  (void)in_sizes; (void)n_in; (void)out_size; (void)ws_size;

  const int*   batch = (const int*)d_in[0];
  const float* feats = (const float*)d_in[1];
  const float* W_ih  = (const float*)d_in[2];
  const float* b_ih  = (const float*)d_in[3];
  const float* W_hh  = (const float*)d_in[4];
  const float* b_hh  = (const float*)d_in[5];
  const float* W_o   = (const float*)d_in[6];
  const float* b_o   = (const float*)d_in[7];
  const float* W_h   = (const float*)d_in[8];
  const float* b_h   = (const float*)d_in[9];
  const float* rmult = (const float*)d_in[10];
  float* out = (float*)d_out;

  // Workspace carve-out (~109 MB total)
  char* ws = (char*)d_ws;
  size_t off = 0;
  auto carve = [&](size_t bytes) -> char* {
    char* p = ws + off;
    off = (off + bytes + 255) & ~(size_t)255;
    return p;
  };
  unsigned short* whh_bf = (unsigned short*)carve((size_t)VDIM * VDIM * 2);
  unsigned short* wo_bf  = (unsigned short*)carve((size_t)VDIM * VDIM * 2);
  unsigned short* wh_bf  = (unsigned short*)carve((size_t)VDIM * VDIM * 2);
  float* xproj  = (float*)carve((size_t)NSTEP * VDIM * 4);
  float* h      = (float*)carve(VDIM * 4);
  float* st     = (float*)carve(VDIM * 4);
  float* o      = (float*)carve(VDIM * 4);
  float* hid    = (float*)carve(VDIM * 4);
  float* blkmax = (float*)carve(NB * 4);
  float* blksum = (float*)carve(NB * 4);
  unsigned* syn = (unsigned*)carve(256);

  init_state<<<16, BS, 0, stream>>>(h, syn, blkmax, blksum);

  const int n4 = VDIM * VDIM / 4;
  f32_to_bf16<<<n4 / BS, BS, 0, stream>>>(W_hh, whh_bf, n4);
  f32_to_bf16<<<n4 / BS, BS, 0, stream>>>(W_o,  wo_bf,  n4);
  f32_to_bf16<<<n4 / BS, BS, 0, stream>>>(W_h,  wh_bf,  n4);

  // 8192 tiles, 8 waves/block -> 1024 blocks
  xproj_gemm<<<1024, BS, 0, stream>>>(batch, feats, W_ih, b_ih, xproj);

  rnn_scan<<<NB, BS, 0, stream>>>(whh_bf, wo_bf, wh_bf, b_hh, b_o, b_h,
                                  xproj, rmult, h, st, o, hid,
                                  blkmax, blksum, syn,
                                  out, out + (size_t)SLEN * VDIM);
}
